// UnetCageGen_9758165696593
// MI455X (gfx1250) — matrix operand order repaired and checked by reading, not compiled, not run on hardware
//
#include <hip/hip_runtime.h>
#include <hip/hip_bf16.h>

// ---------------------------------------------------------------------------
// UnetCageGen forward for MI455X (gfx1250).
// KNN interp in fp32 VALU (LDS-tiled), fold MLP as f16 WMMA GEMMs with f32
// accumulate (v_wmma_f32_16x16x32_f16). Only fold1 is live in the reference.
// ---------------------------------------------------------------------------

#define BATCH 4
#define NQ    8192          // template points per batch
#define K0PAD 480           // 448 feat + 3 prim, padded to mult of 32
#define K1PAD 256           // 224 + 3 prim, padded
#define K2PAD 128           // 112 + 3 prim, padded
#define M0    224
#define M1    112
#define PFROWS 115
#define SLOPE 0.01f

typedef __attribute__((ext_vector_type(16))) _Float16 v16h;
typedef __attribute__((ext_vector_type(8)))  _Float16 v8h;
typedef __attribute__((ext_vector_type(8)))  float    v8f;

union V16U { v16h v; v8h h[2]; };

// ---------------------------------------------------------------------------
__global__ void zero_f32_kernel(float* p, int n) {
    int i = blockIdx.x * blockDim.x + threadIdx.x;
    if (i < n) p[i] = 0.0f;
}

// ---------------------------------------------------------------------------
// Top-3 nearest neighbors per query + per-batch sum of min distances.
// tmpl: (B,3,NQ) channel-major.  pts: (B,Np,3).  256 threads, 1 query/thread.
__global__ void knn_top3_kernel(const float* __restrict__ tmpl,
                                const float* __restrict__ pts, int Np,
                                float* __restrict__ dist3,   // (B,NQ,3)
                                int*   __restrict__ idx3,    // (B,NQ,3)
                                float* __restrict__ hsum)    // (B)
{
    __shared__ float s_pts[512 * 3];
    __shared__ float s_red[256];
    const int tid = threadIdx.x;
    const int bpb = NQ / 256;
    const int b   = blockIdx.x / bpb;
    const int q   = (blockIdx.x % bpb) * 256 + tid;

    const float qx = tmpl[((size_t)b * 3 + 0) * NQ + q];
    const float qy = tmpl[((size_t)b * 3 + 1) * NQ + q];
    const float qz = tmpl[((size_t)b * 3 + 2) * NQ + q];

    float b0 = 3.4e38f, b1 = 3.4e38f, b2 = 3.4e38f;
    int   j0 = 0, j1 = 0, j2 = 0;

    for (int t0 = 0; t0 < Np; t0 += 512) {
        __syncthreads();
        for (int j = tid; j < 512 * 3; j += 256)
            s_pts[j] = pts[((size_t)b * Np + t0) * 3 + j];
        __syncthreads();
        #pragma unroll 4
        for (int i = 0; i < 512; ++i) {
            float dx = qx - s_pts[i * 3 + 0];
            float dy = qy - s_pts[i * 3 + 1];
            float dz = qz - s_pts[i * 3 + 2];
            float dd = dx * dx + dy * dy + dz * dz;
            int   pi = t0 + i;
            if (dd < b0)      { b2 = b1; j2 = j1; b1 = b0; j1 = j0; b0 = dd; j0 = pi; }
            else if (dd < b1) { b2 = b1; j2 = j1; b1 = dd; j1 = pi; }
            else if (dd < b2) { b2 = dd; j2 = pi; }
        }
    }
    size_t o = ((size_t)b * NQ + q) * 3;
    dist3[o] = b0; dist3[o + 1] = b1; dist3[o + 2] = b2;
    idx3[o] = j0;  idx3[o + 1] = j1;  idx3[o + 2] = j2;

    s_red[tid] = b0;
    __syncthreads();
    for (int s = 128; s > 0; s >>= 1) {
        if (tid < s) s_red[tid] += s_red[tid + s];
        __syncthreads();
    }
    if (tid == 0) atomicAdd(&hsum[b], s_red[0]);
}

// ---------------------------------------------------------------------------
// exp-weighted 3-NN feature gather -> f16 feature buffer (B,NQ,K0PAD) at chOff.
__global__ void interp_gather_kernel(const float* __restrict__ dist3,
                                     const int*   __restrict__ idx3,
                                     const float* __restrict__ feats, // (B,C,Np)
                                     int C, int Np, int chOff,
                                     const float* __restrict__ hsum,
                                     _Float16* __restrict__ feat16)
{
    const int tid = threadIdx.x;
    const int bpb = NQ / 256;
    const int b   = blockIdx.x / bpb;
    const int n   = (blockIdx.x % bpb) * 256 + tid;

    const float h   = hsum[b] * (1.0f / (float)NQ) + 1e-8f;
    const float inv = -1.0f / (0.5f * h);
    size_t o = ((size_t)b * NQ + n) * 3;
    float w0 = __expf(dist3[o]     * inv);
    float w1 = __expf(dist3[o + 1] * inv);
    float w2 = __expf(dist3[o + 2] * inv);
    float r  = 1.0f / (w0 + w1 + w2);
    w0 *= r; w1 *= r; w2 *= r;
    int i0 = idx3[o], i1 = idx3[o + 1], i2 = idx3[o + 2];

    const float* fb = feats + (size_t)b * C * Np;
    _Float16* out = feat16 + ((size_t)b * NQ + n) * K0PAD + chOff;
    for (int c = 0; c < C; ++c) {
        const float* fc = fb + (size_t)c * Np;
        out[c] = (_Float16)(w0 * fc[i0] + w1 * fc[i1] + w2 * fc[i2]);
    }
}

// ---------------------------------------------------------------------------
// Fill global-code broadcast, prim channels + zero padding in all layer
// buffers, and the prim rows of point_feat output.
__global__ void fill_aux_kernel(const float* __restrict__ tmpl,   // (B,3,NQ)
                                const float* __restrict__ gcode,  // (B,256)
                                _Float16* __restrict__ feat16,    // (B,NQ,K0PAD)
                                _Float16* __restrict__ buf1,      // (B,NQ,K1PAD)
                                _Float16* __restrict__ buf2,      // (B,NQ,K2PAD)
                                float* __restrict__ pf)           // (B,115,NQ)
{
    const int tid = threadIdx.x;
    const int bpb = NQ / 256;
    const int b   = blockIdx.x / bpb;
    const int n   = (blockIdx.x % bpb) * 256 + tid;

    float t0 = tmpl[((size_t)b * 3 + 0) * NQ + n];
    float t1 = tmpl[((size_t)b * 3 + 1) * NQ + n];
    float t2 = tmpl[((size_t)b * 3 + 2) * NQ + n];

    _Float16* f = feat16 + ((size_t)b * NQ + n) * K0PAD;
    const float* g = gcode + (size_t)b * 256;
    for (int c = 0; c < 256; ++c) f[192 + c] = (_Float16)g[c];
    f[448] = (_Float16)t0; f[449] = (_Float16)t1; f[450] = (_Float16)t2;
    for (int c = 451; c < K0PAD; ++c) f[c] = (_Float16)0.0f;

    _Float16* y1 = buf1 + ((size_t)b * NQ + n) * K1PAD;
    y1[224] = (_Float16)t0; y1[225] = (_Float16)t1; y1[226] = (_Float16)t2;
    for (int c = 227; c < K1PAD; ++c) y1[c] = (_Float16)0.0f;

    _Float16* y2 = buf2 + ((size_t)b * NQ + n) * K2PAD;
    y2[112] = (_Float16)t0; y2[113] = (_Float16)t1; y2[114] = (_Float16)t2;
    for (int c = 115; c < K2PAD; ++c) y2[c] = (_Float16)0.0f;

    pf[((size_t)b * PFROWS + 112) * NQ + n] = t0;
    pf[((size_t)b * PFROWS + 113) * NQ + n] = t1;
    pf[((size_t)b * PFROWS + 114) * NQ + n] = t2;
}

// ---------------------------------------------------------------------------
// fp32 (rows,cols) -> f16 (rowsP,colsP) zero-padded, row-major.
__global__ void pad_weights_kernel(const float* __restrict__ W, int rows, int cols,
                                   _Float16* __restrict__ Wp, int colsP, int total)
{
    int idx = blockIdx.x * 256 + threadIdx.x;
    if (idx >= total) return;
    int r = idx / colsP, c = idx % colsP;
    float v = (r < rows && c < cols) ? W[(size_t)r * cols + c] : 0.0f;
    Wp[idx] = (_Float16)v;
}

// ---------------------------------------------------------------------------
// WMMA GEMM: D(M,N) = W(M,KPAD) * X(N,KPAD)^T per batch.
// block = 128 threads = 4 waves; wave computes one 16x16 tile.
// A layout: lane%16 = row M; lanes<16 hold K {0..7,16..23}, lanes>=16 {8..15,24..31}.
// B layout: lane%16 = col N; lanes<16 hold K {0..7,16..23}, lanes>=16 {8..15,24..31}.
// MODE 0: bias+lrelu -> f16 Y.   MODE 1: same + fp32 point_feat rows.
// MODE 2: bias, residual add with template -> xyz (rows<3 only).
template <int KPAD, int KNEXT, int MODE>
__global__ void gemm_wmma_kernel(const _Float16* __restrict__ Wp,   // (Mpad,KPAD)
                                 const float* __restrict__ bias,
                                 const _Float16* __restrict__ X,    // (B,NQ,KPAD)
                                 _Float16* __restrict__ Y,          // (B,NQ,KNEXT)
                                 float* __restrict__ PF,            // (B,115,NQ)
                                 const float* __restrict__ T,       // (B,3,NQ)
                                 float* __restrict__ XYZ)           // (B,3,NQ)
{
    const int lane   = threadIdx.x & 31;
    const int wave   = threadIdx.x >> 5;
    const int tileN  = (blockIdx.x * 4 + wave) * 16;
    const int tileM  = blockIdx.y * 16;
    const int b      = blockIdx.z;
    const int col    = lane & 15;
    const int rowOff = (lane >> 4) * 8;
    const int n      = tileN + col;
    const int aSel   = (lane < 16) ? 0 : 8;
    const int bSel   = (lane < 16) ? 0 : 16;

    const _Float16* Wrow = Wp + (size_t)(tileM + col) * KPAD;
    const _Float16* Xrow = X + ((size_t)b * NQ + n) * KPAD;

    v8f acc = {};
    #pragma unroll 2
    for (int k0 = 0; k0 < KPAD; k0 += 32) {
        V16U a, bf;
        a.h[0]  = *(const v8h*)(Wrow + k0 + aSel);
        a.h[1]  = *(const v8h*)(Wrow + k0 + 16 + aSel);
        bf.h[0] = *(const v8h*)(Xrow + k0 + aSel);
        bf.h[1] = *(const v8h*)(Xrow + k0 + 16 + aSel);
        acc = __builtin_amdgcn_wmma_f32_16x16x32_f16(
            false, a.v, false, bf.v, (short)0, acc, false, false);
    }

    const int m0 = tileM + rowOff;
    if constexpr (MODE == 2) {
        #pragma unroll
        for (int v = 0; v < 8; ++v) {
            int m = m0 + v;
            if (m < 3) {
                float y = acc[v] + bias[m];
                XYZ[((size_t)b * 3 + m) * NQ + n] =
                    T[((size_t)b * 3 + m) * NQ + n] + y;
            }
        }
    } else {
        v8h out;
        #pragma unroll
        for (int v = 0; v < 8; ++v) {
            int m = m0 + v;
            float y = acc[v] + bias[m];
            y = (y > 0.0f) ? y : SLOPE * y;      // LeakyReLU
            out[v] = (_Float16)y;
            if constexpr (MODE == 1) {
                PF[((size_t)b * PFROWS + m) * NQ + n] = y;   // m < 112 always
            }
        }
        *(v8h*)(Y + ((size_t)b * NQ + n) * KNEXT + m0) = out;
    }
}

// ---------------------------------------------------------------------------
extern "C" void kernel_launch(void* const* d_in, const int* in_sizes, int n_in,
                              void* d_out, int out_size, void* d_ws, size_t ws_size,
                              hipStream_t stream)
{
    const float* tmpl  = (const float*)d_in[0];   // (4,3,8192)
    const float* xyz0  = (const float*)d_in[1];   // (4,4096,3)
    const float* feat0 = (const float*)d_in[2];   // (4,64,4096)
    const float* xyz1  = (const float*)d_in[3];   // (4,1024,3)
    const float* feat1 = (const float*)d_in[4];   // (4,128,1024)
    const float* feat2 = (const float*)d_in[5];   // (4,256)
    // fold0 (d_in[6..11]) is dead in the reference; only fold1 is live.
    const float* W0 = (const float*)d_in[12];     // (224,451)
    const float* b0 = (const float*)d_in[13];
    const float* W1 = (const float*)d_in[14];     // (112,227)
    const float* b1 = (const float*)d_in[15];
    const float* W2 = (const float*)d_in[16];     // (3,115)
    const float* b2 = (const float*)d_in[17];

    float* xyz_out = (float*)d_out;                          // (4,3,8192)
    float* pf_out  = xyz_out + (size_t)BATCH * 3 * NQ;       // (4,115,8192)

    // ---- workspace carve-out (~59 MB) ----
    char*  ws  = (char*)d_ws;
    size_t off = 0;
    auto take = [&](size_t bytes) -> char* {
        char* p = ws + off;
        off += (bytes + 255) & ~(size_t)255;
        return p;
    };
    float*     hsum0  = (float*)take(BATCH * sizeof(float));
    float*     hsum1  = (float*)take(BATCH * sizeof(float));
    float*     dist0  = (float*)take((size_t)BATCH * NQ * 3 * sizeof(float));
    int*       idx0   = (int*)  take((size_t)BATCH * NQ * 3 * sizeof(int));
    float*     dist1  = (float*)take((size_t)BATCH * NQ * 3 * sizeof(float));
    int*       idx1   = (int*)  take((size_t)BATCH * NQ * 3 * sizeof(int));
    _Float16*  feat16 = (_Float16*)take((size_t)BATCH * NQ * K0PAD * 2);
    _Float16*  buf1   = (_Float16*)take((size_t)BATCH * NQ * K1PAD * 2);
    _Float16*  buf2   = (_Float16*)take((size_t)BATCH * NQ * K2PAD * 2);
    _Float16*  W0p    = (_Float16*)take((size_t)M0 * K0PAD * 2);
    _Float16*  W1p    = (_Float16*)take((size_t)M1 * K1PAD * 2);
    _Float16*  W2p    = (_Float16*)take((size_t)16 * K2PAD * 2);

    const int qblocks = BATCH * NQ / 256;   // 128

    // 1) zero bandwidth accumulators (re-done each call: deterministic)
    zero_f32_kernel<<<1, 32, 0, stream>>>(hsum0, BATCH * 2);  // hsum0+hsum1 contiguous? no — zero both:
    zero_f32_kernel<<<1, 32, 0, stream>>>(hsum1, BATCH);

    // 2) weight padding fp32 -> f16
    {
        int t0 = M0 * K0PAD;
        pad_weights_kernel<<<(t0 + 255) / 256, 256, 0, stream>>>(W0, M0, 451, W0p, K0PAD, t0);
        int t1 = M1 * K1PAD;
        pad_weights_kernel<<<(t1 + 255) / 256, 256, 0, stream>>>(W1, M1, 227, W1p, K1PAD, t1);
        int t2 = 16 * K2PAD;
        pad_weights_kernel<<<(t2 + 255) / 256, 256, 0, stream>>>(W2, 3, 115, W2p, K2PAD, t2);
    }

    // 3) KNN top-3 for both levels
    knn_top3_kernel<<<qblocks, 256, 0, stream>>>(tmpl, xyz0, 4096, dist0, idx0, hsum0);
    knn_top3_kernel<<<qblocks, 256, 0, stream>>>(tmpl, xyz1, 1024, dist1, idx1, hsum1);

    // 4) exp-weighted feature gather into feat16 channels [0,64) and [64,192)
    interp_gather_kernel<<<qblocks, 256, 0, stream>>>(dist0, idx0, feat0, 64, 4096, 0,  hsum0, feat16);
    interp_gather_kernel<<<qblocks, 256, 0, stream>>>(dist1, idx1, feat1, 128, 1024, 64, hsum1, feat16);

    // 5) global code broadcast + prim channels + zero padding + pf prim rows
    fill_aux_kernel<<<qblocks, 256, 0, stream>>>(tmpl, feat2, feat16, buf1, buf2, pf_out);

    // 6) fold1 MLP via WMMA
    gemm_wmma_kernel<K0PAD, K1PAD, 0><<<dim3(NQ / 64, M0 / 16, BATCH), 128, 0, stream>>>(
        W0p, b0, feat16, buf1, nullptr, nullptr, nullptr);
    gemm_wmma_kernel<K1PAD, K2PAD, 1><<<dim3(NQ / 64, M1 / 16, BATCH), 128, 0, stream>>>(
        W1p, b1, buf1, buf2, pf_out, nullptr, nullptr);
    gemm_wmma_kernel<K2PAD, K2PAD, 2><<<dim3(NQ / 64, 1, BATCH), 128, 0, stream>>>(
        W2p, b2, buf2, nullptr, nullptr, tmpl, xyz_out);
}
